// Cross_Frequency_Enhanced_Block_59373627900515
// MI455X (gfx1250) — matrix operand order repaired and checked by reading, not compiled, not run on hardware
//
#include <hip/hip_runtime.h>
#include <math.h>

// ---------------------------------------------------------------------------
// Cross_Frequency_Enhanced_Block for MI455X (gfx1250, wave32, WMMA bf16)
// B=32, L=1024, D=512, H=1, E=512, MODES=32, MA_K=128
// Heavy GEMMs -> v_wmma_f32_16x16x32_bf16 with double-buffered LDS pipeline;
// spectral glue (complex tanh attention, mode einsums, moving average, BN)
// on VALU.
// ---------------------------------------------------------------------------

typedef __attribute__((ext_vector_type(16))) __bf16 v16bf;
typedef __attribute__((ext_vector_type(8)))  __bf16 v8bf;
typedef __attribute__((ext_vector_type(8)))  float  v8f;

union BfVec { v16bf v; v8bf h[2]; };

#define CFE_B     32
#define CFE_L     1024
#define CFE_D     512
#define CFE_E     512
#define CFE_MODES 32
#define TWO_PI_OVER_L 0.006135923151542565f  // 2*pi/1024

// ---------------------------------------------------------------------------
// Generic WMMA GEMM:  C[M,N] = A[M,K] (bf16, row-major, lda) @ Bt[N,K]^T
// Bt is stored (N,K) row-major (i.e. weights as (out,in)), bf16.
// Block: 256 threads = 8 waves; tile 128(M) x 64(N); K-step 32.
// Double-buffered LDS, one barrier per K-step; all 5 fragments (A + 4xB)
// hoisted before the 4-WMMA chain so the XDL pipe issues back-to-back.
// OUT_MODE: 0 = f32 row-major (Cf,ldc), 1 = bf16 row-major (Cb,ldc),
//           2 = bf16 transposed-per-batch: dst = (m/transR)*transR*transN
//                                              + n*transR + (m%transR)
// ---------------------------------------------------------------------------
template<bool HAS_BIAS, bool HAS_RESID, bool DO_GELU, bool HAS_ROWAFF, int OUT_MODE>
__global__ __launch_bounds__(256)
void wmma_gemm_kernel(const __bf16* __restrict__ A, int lda,
                      const __bf16* __restrict__ Bt, int ldb,
                      int M, int N, int K,
                      const float* __restrict__ bias,
                      const float* __restrict__ resid, int ldr,
                      const float* __restrict__ rowscale,
                      const float* __restrict__ rowshift, int rowmod,
                      float* __restrict__ Cf, __bf16* __restrict__ Cb, int ldc,
                      int transR, int transN)
{
    // 40-element (80B) row stride keeps the 16B ds_load_b128 lanes on
    // disjoint bank quadruples.
    __shared__ __bf16 sA[2][128][40];
    __shared__ __bf16 sB[2][64][40];

    const int tid  = threadIdx.x;
    const int wave = tid >> 5;
    const int lane = tid & 31;
    const int lm   = lane & 15;      // column / row-in-half selector
    const int lh   = lane >> 4;      // half selector
    const int m0   = blockIdx.y * 128;
    const int n0   = blockIdx.x * 64;

    // A staging: 128 rows x 2 halves of 16 elems  (all 256 threads, 32B each)
    const int arow  = tid >> 1;
    const int ahalf = tid & 1;
    // B staging: 64 rows x 4 quarters of 8 elems (all 256 threads, 16B each)
    const int brow  = tid >> 2;
    const int bq4   = tid & 3;

    const __bf16* aptr = A  + (size_t)(m0 + arow) * lda + ahalf * 16;
    const __bf16* bptr = Bt + (size_t)(n0 + brow) * ldb + bq4 * 8;

    v8f acc[4] = {};

    // ---- prologue: stage K-tile 0 into buffer 0 ----
    BfVec ta;
    v8bf  tb;
    ta.v = *(const v16bf*)(aptr);
    tb   = *(const v8bf*)(bptr);
    *(v8bf*)(&sA[0][arow][ahalf * 16])     = ta.h[0];
    *(v8bf*)(&sA[0][arow][ahalf * 16 + 8]) = ta.h[1];
    *(v8bf*)(&sB[0][brow][bq4 * 8])        = tb;
    __syncthreads();

    for (int k0 = 0; k0 < K; k0 += 32) {
        const int  buf  = (k0 >> 5) & 1;
        const bool more = (k0 + 32) < K;

        // issue global loads for the next tile early (hidden behind WMMAs)
        if (more) {
            ta.v = *(const v16bf*)(aptr + k0 + 32);
            tb   = *(const v8bf*)(bptr + k0 + 32);
            if (k0 + 64 < K)
                __builtin_prefetch(aptr + k0 + 64, 0, 1);
        }

        // A fragment (16x32 bf16): lanes 0-15 -> K half 0, lanes 16-31 -> K half 8
        BfVec af, bfr[4];
        af.h[0] = *(const v8bf*)(&sA[buf][wave * 16 + lm][lh * 8]);
        af.h[1] = *(const v8bf*)(&sA[buf][wave * 16 + lm][16 + lh * 8]);
#pragma unroll
        for (int c = 0; c < 4; ++c) {
            // B fragment (32x16 bf16): lane n=lm, K = lh*8 and 16 + lh*8
            bfr[c].h[0] = *(const v8bf*)(&sB[buf][c * 16 + lm][lh * 8]);
            bfr[c].h[1] = *(const v8bf*)(&sB[buf][c * 16 + lm][16 + lh * 8]);
        }
#pragma unroll
        for (int c = 0; c < 4; ++c)
            acc[c] = __builtin_amdgcn_wmma_f32_16x16x32_bf16(
                false, af.v, false, bfr[c].v, (short)0, acc[c], false, false);

        // stage next tile into the alternate buffer
        if (more) {
            *(v8bf*)(&sA[buf ^ 1][arow][ahalf * 16])     = ta.h[0];
            *(v8bf*)(&sA[buf ^ 1][arow][ahalf * 16 + 8]) = ta.h[1];
            *(v8bf*)(&sB[buf ^ 1][brow][bq4 * 8])        = tb;
        }
        __syncthreads();
    }

#pragma unroll
    for (int c = 0; c < 4; ++c) {
        const int ng = n0 + c * 16 + lm;
        float bv = 0.f;
        if (HAS_BIAS) bv = bias[ng];
#pragma unroll
        for (int r = 0; r < 8; ++r) {
            const int mg = m0 + wave * 16 + lh * 8 + r;  // C layout: VGPR r -> M=r+8*lh
            float v = acc[c][r] + bv;
            if (HAS_RESID)  v += resid[(size_t)mg * ldr + ng];
            if (DO_GELU)    v = 0.5f * v * (1.f + erff(v * 0.70710678118654752f));
            if (HAS_ROWAFF) { int l = mg % rowmod; v = v * rowscale[l] + rowshift[l]; }
            if (OUT_MODE == 0) {
                Cf[(size_t)mg * ldc + ng] = v;
            } else if (OUT_MODE == 1) {
                Cb[(size_t)mg * ldc + ng] = (__bf16)v;
            } else {
                size_t d = (size_t)(mg / transR) * ((size_t)transR * transN)
                         + (size_t)ng * transR + (size_t)(mg % transR);
                Cb[d] = (__bf16)v;
            }
        }
    }
}

// ---------------------------------------------------------------------------
// Small helper kernels
// ---------------------------------------------------------------------------
__global__ void f32_to_bf16_kernel(const float* __restrict__ src,
                                   __bf16* __restrict__ dst, int n)
{
    int i = blockIdx.x * 256 + threadIdx.x;
    if (i < n) dst[i] = (__bf16)src[i];
}

// forward DFT basis (64 x 1024): rows 0..31 = cos(2*pi*m*l/L); rows 32..63 = -sin
__global__ void fill_basis_fwd_kernel(__bf16* __restrict__ bf)
{
    int id = blockIdx.x * 256 + threadIdx.x;  // 65536
    int n = id >> 10, l = id & 1023;
    int m = n & 31;
    float ph = (float)((m * l) & 1023) * TWO_PI_OVER_L;
    float v = (n < 32) ? __cosf(ph) : -__sinf(ph);
    bf[id] = (__bf16)v;
}

// inverse DFT basis (1024 x 64), scale 1/(L*D*D) and the m==0 / 2x factors folded
__global__ void fill_basis_inv_kernel(__bf16* __restrict__ bi)
{
    int id = blockIdx.x * 256 + threadIdx.x;  // 65536
    int l = id >> 6, m = id & 63;
    int mm = m & 31;
    float c = ((mm == 0) ? 1.f : 2.f) * 3.7252902984619140625e-9f;  // 1/(1024*512*512)
    float ph = (float)((mm * l) & 1023) * TWO_PI_OVER_L;
    float v = (m < 32) ? c * __cosf(ph) : -c * __sinf(ph);
    bi[id] = (__bf16)v;
}

__global__ void bn_coef_kernel(const float* __restrict__ gamma,
                               const float* __restrict__ beta,
                               const float* __restrict__ mean,
                               const float* __restrict__ var,
                               float* __restrict__ sc, float* __restrict__ sh)
{
    int l = blockIdx.x * 256 + threadIdx.x;
    if (l < CFE_L) {
        float inv = gamma[l] * rsqrtf(var[l] + 1e-5f);
        sc[l] = inv;
        sh[l] = beta[l] - mean[l] * inv;
    }
}

// xqk[b,x,y] = tanh_c( sum_e XQ[b,e,x] * XK[b,e,y] )   (complex, no conj)
__global__ __launch_bounds__(256)
void xqk_tanh_kernel(const float* __restrict__ XQ, const float* __restrict__ XK,
                     float2* __restrict__ xqk)
{
    const int b = blockIdx.x >> 5;
    const int x = blockIdx.x & 31;
    const int tid = threadIdx.x;
    const int y = tid & 31, part = tid >> 5;   // 8 partial sums over e

    float sr = 0.f, si = 0.f;
    for (int e = part * 64; e < part * 64 + 64; ++e) {
        const float* q = XQ + ((size_t)b * CFE_E + e) * 64;
        const float* k = XK + ((size_t)b * CFE_E + e) * 64;
        float qr = q[x], qi = q[32 + x];
        float kr = k[y], ki = k[32 + y];
        sr += qr * kr - qi * ki;
        si += qr * ki + qi * kr;
    }
    __shared__ float rr[256], ri[256];
    rr[tid] = sr; ri[tid] = si;
    __syncthreads();
    if (part == 0) {
        for (int p = 1; p < 8; ++p) { sr += rr[y + 32 * p]; si += ri[y + 32 * p]; }
        // complex tanh: (sinh(2a) + i sin(2b)) / (cosh(2a) + cos(2b))
        float a2 = 2.f * sr, b2 = 2.f * si;
        float tr, ti;
        if (fabsf(a2) > 20.f) {
            tr = (a2 > 0.f) ? 1.f : -1.f; ti = 0.f;
        } else {
            float e = __expf(a2), einv = 1.f / e;
            float sh = 0.5f * (e - einv), ch = 0.5f * (e + einv);
            float den = ch + __cosf(b2);
            tr = sh / den;
            ti = __sinf(b2) / den;
        }
        xqk[((size_t)b * 32 + x) * 32 + y] = make_float2(tr, ti);
    }
}

// xqkv[b,e,x] = sum_y xqk[b,x,y] * XK[b,e,y]
__global__ void xqkv_kernel(const float2* __restrict__ xqk,
                            const float* __restrict__ XK,
                            float2* __restrict__ xqkv)
{
    int id = blockIdx.x * 256 + threadIdx.x;   // B*E*MODES = 524288
    int x = id & 31, e = (id >> 5) & 511, b = id >> 14;
    const float2* row = xqk + ((size_t)b * 32 + x) * 32;
    const float* k = XK + ((size_t)b * CFE_E + e) * 64;
    float sr = 0.f, si = 0.f;
    for (int y = 0; y < 32; ++y) {
        float2 a = row[y];
        float kr = k[y], ki = k[32 + y];
        sr += a.x * kr - a.y * ki;
        si += a.x * ki + a.y * kr;
    }
    xqkv[((size_t)b * CFE_E + e) * 32 + x] = make_float2(sr, si);
}

// xqkvw[b,o,x] = sum_e xqkv[b,e,x] * (wr+ i wi)[e,o,x]  -> bf16 iDFT A matrix
__global__ void xqkvw_kernel(const float2* __restrict__ xqkv,
                             const float* __restrict__ wr,
                             const float* __restrict__ wi,
                             __bf16* __restrict__ Aid)
{
    int id = blockIdx.x * 256 + threadIdx.x;   // B*E*MODES = 524288
    int x = id & 31, o = (id >> 5) & 511, b = id >> 14;
    float sr = 0.f, si = 0.f;
    for (int e = 0; e < CFE_E; ++e) {
        float2 a = xqkv[((size_t)b * CFE_E + e) * 32 + x];
        size_t wo = ((size_t)e * CFE_E + o) * 32 + x;
        float r = wr[wo], m = wi[wo];
        sr += a.x * r - a.y * m;
        si += a.x * m + a.y * r;
    }
    __bf16* dst = Aid + ((size_t)b * CFE_E + o) * 64;
    dst[x] = (__bf16)sr;
    dst[32 + x] = (__bf16)si;
}

// series_decomp residual: x2 = x1 - moving_avg(x1, 128) (edge-replicate window
// [l-64, l+63]); writes fp32 in place and a bf16 copy. Block = one (b,d) row.
__global__ __launch_bounds__(256)
void moving_avg_kernel(const float* __restrict__ x1,
                       float* __restrict__ x2, __bf16* __restrict__ x2b)
{
    const int b = blockIdx.x >> 9;     // /512
    const int d = blockIdx.x & 511;
    __shared__ float row[CFE_L];
    for (int l = threadIdx.x; l < CFE_L; l += 256)
        row[l] = x1[((size_t)b * CFE_L + l) * CFE_D + d];
    __syncthreads();
    for (int l = threadIdx.x; l < CFE_L; l += 256) {
        float s = 0.f;
        for (int j = -64; j < 64; ++j) {
            int idx = l + j;
            idx = (idx < 0) ? 0 : ((idx > CFE_L - 1) ? CFE_L - 1 : idx);
            s += row[idx];
        }
        float v = row[l] - s * (1.f / 128.f);
        size_t off = ((size_t)b * CFE_L + l) * CFE_D + d;
        x2[off] = v;
        x2b[off] = (__bf16)v;
    }
}

// ---------------------------------------------------------------------------
// Launch
// ---------------------------------------------------------------------------
extern "C" void kernel_launch(void* const* d_in, const int* in_sizes, int n_in,
                              void* d_out, int out_size, void* d_ws, size_t ws_size,
                              hipStream_t stream)
{
    const float* x      = (const float*)d_in[0];
    const float* Wq     = (const float*)d_in[1];
    const float* bq     = (const float*)d_in[2];
    const float* Wk     = (const float*)d_in[3];
    const float* bk     = (const float*)d_in[4];
    const float* Wo     = (const float*)d_in[5];
    const float* bo     = (const float*)d_in[6];
    const float* w_real = (const float*)d_in[7];
    const float* w_imag = (const float*)d_in[8];
    const float* c1w    = (const float*)d_in[9];
    const float* c2w    = (const float*)d_in[10];
    const float* bng    = (const float*)d_in[11];
    const float* bnb    = (const float*)d_in[12];
    const float* bnm    = (const float*)d_in[13];
    const float* bnv    = (const float*)d_in[14];
    float* out = (float*)d_out;

    char* ws = (char*)d_ws;
    constexpr size_t MB = 1ull << 20;
    // Aliased workspace layout (~180 MB):
    //   [0,32M)   xb (bf16 x)          -> reused as fre_t after DFT stage
    //   [32,96M)  q_t | k_t (bf16)     -> reused as x1/x2 (fp32 64MB)
    __bf16* xb    = (__bf16*)(ws + 0);
    __bf16* fre_t = (__bf16*)(ws + 0);
    __bf16* q_t   = (__bf16*)(ws + 32 * MB);
    __bf16* k_t   = (__bf16*)(ws + 64 * MB);
    float*  x1    = (float*) (ws + 32 * MB);   // aliases q_t/k_t (dead by then)
    float*  x2    = x1;                        // in-place decomp
    float*  XQ    = (float*) (ws + 96 * MB);
    float*  XK    = (float*) (ws + 100 * MB);
    float2* xqk   = (float2*)(ws + 104 * MB);
    float2* xqkv  = (float2*)(ws + 105 * MB);
    __bf16* Aid   = (__bf16*)(ws + 109 * MB);
    __bf16* x2b   = (__bf16*)(ws + 112 * MB);
    __bf16* y1    = (__bf16*)(ws + 144 * MB);
    __bf16* basF  = (__bf16*)(ws + 176 * MB);               // 64x1024
    __bf16* basI  = (__bf16*)(ws + 176 * MB + 131072);      // 1024x64
    float*  bnsc  = (float*) (ws + 176 * MB + 262144);
    float*  bnsh  = (float*) (ws + 176 * MB + 262144 + 4096);
    __bf16* Wqb   = (__bf16*)(ws + 177 * MB);
    __bf16* Wkb   = (__bf16*)(ws + 177 * MB + 1 * 524288);
    __bf16* Wob   = (__bf16*)(ws + 177 * MB + 2 * 524288);
    __bf16* C1b   = (__bf16*)(ws + 177 * MB + 3 * 524288);
    __bf16* C2b   = (__bf16*)(ws + 177 * MB + 4 * 524288);

    const int NTOK = CFE_B * CFE_L;            // 32768
    const int NW   = CFE_D * CFE_D;            // 262144

    // ---- precision conversion + constants ----
    f32_to_bf16_kernel<<<(NTOK * CFE_D) / 256, 256, 0, stream>>>(x, xb, NTOK * CFE_D);
    f32_to_bf16_kernel<<<NW / 256, 256, 0, stream>>>(Wq, Wqb, NW);
    f32_to_bf16_kernel<<<NW / 256, 256, 0, stream>>>(Wk, Wkb, NW);
    f32_to_bf16_kernel<<<NW / 256, 256, 0, stream>>>(Wo, Wob, NW);
    f32_to_bf16_kernel<<<NW / 256, 256, 0, stream>>>(c1w, C1b, NW);
    f32_to_bf16_kernel<<<NW / 256, 256, 0, stream>>>(c2w, C2b, NW);
    fill_basis_fwd_kernel<<<256, 256, 0, stream>>>(basF);
    fill_basis_inv_kernel<<<256, 256, 0, stream>>>(basI);
    bn_coef_kernel<<<4, 256, 0, stream>>>(bng, bnb, bnm, bnv, bnsc, bnsh);

    // ---- q/k projections: (32768,512)x(512,512), store transposed (B,E,L) bf16
    wmma_gemm_kernel<true, false, false, false, 2>
        <<<dim3(CFE_D / 64, NTOK / 128), 256, 0, stream>>>(
        xb, CFE_D, Wqb, CFE_D, NTOK, CFE_D, CFE_D,
        bq, nullptr, 0, nullptr, nullptr, 0,
        nullptr, q_t, 0, CFE_L, CFE_D);
    wmma_gemm_kernel<true, false, false, false, 2>
        <<<dim3(CFE_D / 64, NTOK / 128), 256, 0, stream>>>(
        xb, CFE_D, Wkb, CFE_D, NTOK, CFE_D, CFE_D,
        bk, nullptr, 0, nullptr, nullptr, 0,
        nullptr, k_t, 0, CFE_L, CFE_D);

    // ---- forward DFT (32 lowest rfft bins) as GEMM: (B*E,1024)x(1024,64)
    const int BE = CFE_B * CFE_E;              // 16384
    wmma_gemm_kernel<false, false, false, false, 0>
        <<<dim3(1, BE / 128), 256, 0, stream>>>(
        q_t, CFE_L, basF, CFE_L, BE, 64, CFE_L,
        nullptr, nullptr, 0, nullptr, nullptr, 0,
        XQ, nullptr, 64, 0, 0);
    wmma_gemm_kernel<false, false, false, false, 0>
        <<<dim3(1, BE / 128), 256, 0, stream>>>(
        k_t, CFE_L, basF, CFE_L, BE, 64, CFE_L,
        nullptr, nullptr, 0, nullptr, nullptr, 0,
        XK, nullptr, 64, 0, 0);

    // ---- mode-space complex attention ----
    xqk_tanh_kernel<<<CFE_B * CFE_MODES, 256, 0, stream>>>(XQ, XK, xqk);
    xqkv_kernel<<<(CFE_B * CFE_E * CFE_MODES) / 256, 256, 0, stream>>>(xqk, XK, xqkv);
    xqkvw_kernel<<<(CFE_B * CFE_E * CFE_MODES) / 256, 256, 0, stream>>>(
        xqkv, w_real, w_imag, Aid);

    // ---- irfft as GEMM: (B*E,64)x(64,1024), store transposed -> (B,L,E) bf16
    wmma_gemm_kernel<false, false, false, false, 2>
        <<<dim3(CFE_L / 64, BE / 128), 256, 0, stream>>>(
        Aid, 64, basI, 64, BE, CFE_L, 64,
        nullptr, nullptr, 0, nullptr, nullptr, 0,
        nullptr, fre_t, 0, CFE_E, CFE_L);

    // ---- out projection + residual: x1 = x + fre@Wo^T + bo (fp32) ----
    wmma_gemm_kernel<true, true, false, false, 0>
        <<<dim3(CFE_D / 64, NTOK / 128), 256, 0, stream>>>(
        fre_t, CFE_E, Wob, CFE_E, NTOK, CFE_D, CFE_E,
        bo, x, CFE_D, nullptr, nullptr, 0,
        x1, nullptr, CFE_D, 0, 0);

    // ---- series decomposition residual (in place) + bf16 copy ----
    moving_avg_kernel<<<CFE_B * CFE_D, 256, 0, stream>>>(x1, x2, x2b);

    // ---- conv1 (512x512) + exact GELU -> bf16 ----
    wmma_gemm_kernel<false, false, true, false, 1>
        <<<dim3(CFE_D / 64, NTOK / 128), 256, 0, stream>>>(
        x2b, CFE_D, C1b, CFE_D, NTOK, CFE_D, CFE_D,
        nullptr, nullptr, 0, nullptr, nullptr, 0,
        nullptr, y1, CFE_D, 0, 0);

    // ---- conv2 + residual(x2) + BatchNorm over L (row affine) -> d_out ----
    wmma_gemm_kernel<false, true, false, true, 0>
        <<<dim3(CFE_D / 64, NTOK / 128), 256, 0, stream>>>(
        y1, CFE_D, C2b, CFE_D, NTOK, CFE_D, CFE_D,
        nullptr, x2, CFE_D, bnsc, bnsh, CFE_L,
        out, nullptr, CFE_D, 0, 0);
}
